// Block_10496900071540
// MI455X (gfx1250) — compile-verified, matrix operand order checked by reading, not run on hardware
//
#include <hip/hip_runtime.h>
#include <hip/hip_bf16.h>
#include <math.h>

// ---------------------------------------------------------------------------
// Types for CDNA5 WMMA (gfx1250, wave32)
// ---------------------------------------------------------------------------
typedef _Float16 half_t;
typedef __attribute__((ext_vector_type(8)))  _Float16 v8h;
typedef __attribute__((ext_vector_type(16))) _Float16 v16h;
typedef __attribute__((ext_vector_type(8)))  float    v8f;

#define BB   2
#define TT   2048
#define CC   1024
#define HH   16
#define DD   64
#define FFF  4096
#define MM   (BB * TT)          // 4096 rows
#define LNEPS 1e-5f

// ---------------------------------------------------------------------------
// WMMA helpers
// ---------------------------------------------------------------------------
static __device__ __forceinline__ v8f wmma32f16(v16h a, v16h b, v8f c) {
  // D = A(16x32 f16) * B(32x16 f16) + C(16x16 f32)
  return __builtin_amdgcn_wmma_f32_16x16x32_f16(false, a, false, b,
                                                (short)0, c, false, false);
}
static __device__ __forceinline__ v16h cat16(v8h lo, v8h hi) {
  return __builtin_shufflevector(lo, hi, 0,1,2,3,4,5,6,7,
                                         8,9,10,11,12,13,14,15);
}
static __device__ __forceinline__ v8f zero8() {
  v8f z;
#pragma unroll
  for (int i = 0; i < 8; ++i) z[i] = 0.f;
  return z;
}
// Load a 16x32 f16 fragment (A layout; identical per-lane pattern for the B
// fragment when the buffer is stored N-major/K-contiguous).
// ISA layout: lane<16 holds row (lane), K = {koff..koff+7, koff+16..koff+23}
// with koff = 0 (lanes 0-15) or 8 (lanes 16-31); each 8-K run is contiguous.
static __device__ __forceinline__ v16h load_frag(const half_t* base,
                                                 int rowStride /*halves*/) {
  int lane = threadIdx.x & 31;
  int row  = lane & 15;
  int koff = (lane < 16) ? 0 : 8;
  const half_t* p = base + (size_t)row * rowStride + koff;
  v8h lo = *(const v8h*)(p);
  v8h hi = *(const v8h*)(p + 16);
  return cat16(lo, hi);
}
// CDNA5 async global->LDS copy (16B per active lane), tracked by ASYNCcnt.
static __device__ __forceinline__ void async_copy_b128(unsigned lds_byte_addr,
                                                       const void* gptr) {
  asm volatile("global_load_async_to_lds_b128 %0, %1, off"
               :: "v"(lds_byte_addr), "v"(gptr) : "memory");
}

// ---------------------------------------------------------------------------
// Utility kernels
// ---------------------------------------------------------------------------
__global__ void k_zero_f32(float* p, long long n) {
  long long i = (long long)blockIdx.x * blockDim.x + threadIdx.x;
  long long stride = (long long)gridDim.x * blockDim.x;
  for (; i < n; i += stride) p[i] = 0.f;
}

// Transpose + convert: in is K x N fp32 (row-major), out is N x K f16.
// Tiled through LDS so both global streams are coalesced.
__global__ __launch_bounds__(256)
void k_transpose_f32_f16(const float* __restrict__ in,
                         half_t* __restrict__ out, int K, int N) {
  __shared__ float tile[32][33];
  const int k0 = blockIdx.y * 32;
  const int n0 = blockIdx.x * 32;
#pragma unroll
  for (int i = threadIdx.y; i < 32; i += 8)
    tile[i][threadIdx.x] = in[(size_t)(k0 + i) * N + n0 + threadIdx.x];
  __syncthreads();
#pragma unroll
  for (int i = threadIdx.y; i < 32; i += 8)
    out[(size_t)(n0 + i) * K + k0 + threadIdx.x] = (half_t)tile[threadIdx.x][i];
}

// LayerNorm over C=1024, one 128-thread block per row, f16 output for WMMA.
__global__ __launch_bounds__(128)
void k_layernorm_f16(const float* __restrict__ x,
                     const float* __restrict__ g,
                     const float* __restrict__ b,
                     half_t* __restrict__ out) {
  const int row = blockIdx.x;
  const float* xr = x + (size_t)row * CC;
  float s1 = 0.f, s2 = 0.f;
#pragma unroll
  for (int i = threadIdx.x; i < CC; i += 128) {
    float v = xr[i];
    s1 += v; s2 += v * v;
  }
#pragma unroll
  for (int o = 16; o >= 1; o >>= 1) {
    s1 += __shfl_xor(s1, o, 32);
    s2 += __shfl_xor(s2, o, 32);
  }
  __shared__ float red1[4], red2[4];
  int wave = threadIdx.x >> 5;
  if ((threadIdx.x & 31) == 0) { red1[wave] = s1; red2[wave] = s2; }
  __syncthreads();
  s1 = red1[0] + red1[1] + red1[2] + red1[3];
  s2 = red2[0] + red2[1] + red2[2] + red2[3];
  const float mu   = s1 * (1.f / CC);
  const float var  = s2 * (1.f / CC) - mu * mu;
  const float rstd = rsqrtf(var + LNEPS);
  half_t* orow = out + (size_t)row * CC;
  for (int i = threadIdx.x; i < CC; i += 128)
    orow[i] = (half_t)((xr[i] - mu) * rstd * g[i] + b[i]);
}

// ---------------------------------------------------------------------------
// Tiled WMMA GEMM: C[M,N] = act(A[M,K]_f16 * Bt[N,K]_f16^T + bias)(+residual)
//  - B is pre-transposed (N-major, K-contiguous) so both tiles stage as pure
//    16B copies via CDNA5 async global->LDS (ASYNCcnt), double-buffered.
//  - Block tile 128x128, 256 threads = 8 waves (2x4), wave tile 64x32
//    = 4x2 WMMA accumulators, K-step 32.
//  - act: 0 = none, 1 = exact GELU. Optional fp32 / f16 outputs.
//  - M,N multiples of 128; K multiple of 32 (true for all uses here).
// ---------------------------------------------------------------------------
#define LSTR 40  // padded LDS row stride in halves (80B, 16B-aligned)
__global__ __launch_bounds__(256)
void k_gemm_f16(const half_t* __restrict__ A, const half_t* __restrict__ Bt,
                const float* __restrict__ bias,
                const float* __restrict__ resid,
                float* __restrict__ outF, half_t* __restrict__ outH,
                int M, int N, int K, int act) {
  __shared__ half_t lA[2][128 * LSTR];   // A tile (m-local, k)
  __shared__ half_t lB[2][128 * LSTR];   // B tile (n-local, k)
  const int tid  = threadIdx.x;
  const int lane = tid & 31;
  const int wave = tid >> 5;
  const int wm = wave >> 2, wn = wave & 3;   // 2 x 4 wave grid
  const int m0 = blockIdx.y * 128;
  const int n0 = blockIdx.x * 128;

  v8f acc[4][2];
#pragma unroll
  for (int i = 0; i < 4; ++i)
#pragma unroll
    for (int j = 0; j < 2; ++j) acc[i][j] = zero8();

  // Issue one tile's async copies: 4 b128 per thread (2 A + 2 B chunks).
  auto stage = [&](int buf, int k0) {
#pragma unroll
    for (int c = tid; c < 512; c += 256) {
      const int row = c >> 2, kb = (c & 3) * 8;
      async_copy_b128((unsigned)(uintptr_t)&lA[buf][row * LSTR + kb],
                      A + (size_t)(m0 + row) * K + k0 + kb);
      async_copy_b128((unsigned)(uintptr_t)&lB[buf][row * LSTR + kb],
                      Bt + (size_t)(n0 + row) * K + k0 + kb);
    }
  };

  const int nk = K >> 5;
  stage(0, 0);
  for (int kt = 0; kt < nk; ++kt) {
    const int cur = kt & 1;
    if (kt + 1 < nk) {
      stage(cur ^ 1, (kt + 1) << 5);
      // 8 async in flight; current tile's 4 complete first (in-order done).
      asm volatile("s_wait_asynccnt 0x4" ::: "memory");
    } else {
      asm volatile("s_wait_asynccnt 0x0" ::: "memory");
    }
    __syncthreads();                     // current tile visible to all waves

    v16h af[4], bf[2];
#pragma unroll
    for (int f = 0; f < 4; ++f)
      af[f] = load_frag(&lA[cur][(wm * 64 + f * 16) * LSTR], LSTR);
#pragma unroll
    for (int f = 0; f < 2; ++f)
      bf[f] = load_frag(&lB[cur][(wn * 32 + f * 16) * LSTR], LSTR);
#pragma unroll
    for (int i = 0; i < 4; ++i)
#pragma unroll
      for (int j = 0; j < 2; ++j)
        acc[i][j] = wmma32f16(af[i], bf[j], acc[i][j]);

    __syncthreads();                     // done reading before overwrite
  }

  // Epilogue. C layout: VGPR a -> row a (lanes 0-15) / a+8 (lanes 16-31),
  // col = lane&15.
  const int hi = (lane >= 16) ? 8 : 0;
#pragma unroll
  for (int i = 0; i < 4; ++i) {
#pragma unroll
    for (int j = 0; j < 2; ++j) {
      const int col = n0 + wn * 32 + j * 16 + (lane & 15);
      const float bv = bias[col];
#pragma unroll
      for (int a = 0; a < 8; ++a) {
        const int row = m0 + wm * 64 + i * 16 + a + hi;
        float v = acc[i][j][a] + bv;
        if (act == 1) v = 0.5f * v * (1.0f + erff(v * 0.70710678118654752f));
        if (resid) v += resid[(size_t)row * N + col];
        if (outF) outF[(size_t)row * N + col] = v;
        if (outH) outH[(size_t)row * N + col] = (half_t)v;
      }
    }
  }
}

// ---------------------------------------------------------------------------
// Attention: one wave per (b, h, 16-row tile). Two-pass online softmax with
// exact renormalization so att_mean is exact. Q,K,V,Y are f16 (B,T,C) with
// head h occupying columns [h*D, h*D+64). att is fp32 (B,T,T), pre-zeroed,
// accumulated with atomics (mean over H heads).
// ---------------------------------------------------------------------------
__global__ __launch_bounds__(32)
void k_attention(const half_t* __restrict__ q, const half_t* __restrict__ k,
                 const half_t* __restrict__ v, half_t* __restrict__ y,
                 float* __restrict__ att) {
  const int ntiles = TT / 16;
  const int tile = blockIdx.x % ntiles;
  const int bh   = blockIdx.x / ntiles;
  const int b    = bh / HH;
  const int h    = bh % HH;
  const int t0   = tile * 16;
  const int lane = threadIdx.x & 31;
  const int hi8  = (lane >= 16) ? 8 : 0;
  const int lc   = lane & 15;
  const float scale = 0.125f;       // 1/sqrt(64)
  const float invH  = 1.0f / HH;
  const int jend = t0 + 16;         // causal: columns [0, t0+15]

  const half_t* qbase = q + ((size_t)(b * TT + t0)) * CC + h * DD;
  const half_t* kbase = k + ((size_t)b * TT) * CC + h * DD;
  const half_t* vbase = v + ((size_t)b * TT) * CC + h * DD;

  // Q fragments for K-dim = 64 (two chained x32 WMMAs), loaded once.
  const v16h qf0 = load_frag(qbase, CC);
  const v16h qf1 = load_frag(qbase + 32, CC);

  // -------- Pass 1: per-row running max m[] and denom l[] --------
  float m[8], l[8];
#pragma unroll
  for (int a = 0; a < 8; ++a) { m[a] = -INFINITY; l[a] = 0.f; }

  for (int js = 0; js < jend; js += 16) {
    v16h kf0 = load_frag(kbase + (size_t)js * CC, CC);
    v16h kf1 = load_frag(kbase + (size_t)js * CC + 32, CC);
    v8f s = zero8();
    s = wmma32f16(qf0, kf0, s);
    s = wmma32f16(qf1, kf1, s);
#pragma unroll
    for (int a = 0; a < 8; ++a) {
      const int trow = t0 + a + hi8;
      const int jg = js + lc;
      float sv = s[a] * scale;
      if (jg > trow) sv = -INFINITY;
      float rm = sv;
#pragma unroll
      for (int o = 8; o >= 1; o >>= 1) rm = fmaxf(rm, __shfl_xor(rm, o, 32));
      const float nm = fmaxf(m[a], rm);
      float pe = __expf(sv - nm);
      float rs = pe;
#pragma unroll
      for (int o = 8; o >= 1; o >>= 1) rs += __shfl_xor(rs, o, 32);
      l[a] = l[a] * __expf(m[a] - nm) + rs;
      m[a] = nm;
    }
  }
  float linv[8];
#pragma unroll
  for (int a = 0; a < 8; ++a) linv[a] = 1.0f / l[a];

  // -------- Pass 2: P = exp(S-m)/l ; att_mean += P/H ; Y += P @ V --------
  __shared__ half_t pA[16 * 32];    // P tile as WMMA-A operand (row, k=j)
  __shared__ half_t vt[64 * 32];    // V^T tile: (d, j-local)
  v8f yacc[4];
#pragma unroll
  for (int dt = 0; dt < 4; ++dt) yacc[dt] = zero8();

  for (int j0 = 0; j0 < jend; j0 += 32) {
#pragma unroll
    for (int sub = 0; sub < 2; ++sub) {
      const int js = j0 + sub * 16;
      if (js < jend) {
        v16h kf0 = load_frag(kbase + (size_t)js * CC, CC);
        v16h kf1 = load_frag(kbase + (size_t)js * CC + 32, CC);
        v8f s = zero8();
        s = wmma32f16(qf0, kf0, s);
        s = wmma32f16(qf1, kf1, s);
#pragma unroll
        for (int a = 0; a < 8; ++a) {
          const int trow = t0 + a + hi8;
          const int jg = js + lc;
          float pv = 0.f;
          if (jg <= trow) {
            pv = __expf(s[a] * scale - m[a]) * linv[a];
            atomicAdd(&att[((size_t)(b * TT + trow)) * TT + jg], pv * invH);
          }
          pA[(a + hi8) * 32 + sub * 16 + lc] = (half_t)pv;
        }
      } else {
#pragma unroll
        for (int a = 0; a < 8; ++a)
          pA[(a + hi8) * 32 + sub * 16 + lc] = (half_t)0.f;
      }
    }
    // Stage V^T: lane s loads one V row (64 contiguous halves), scatter to vt.
    const int sg = j0 + lane;
    if (sg < TT) {
      const half_t* vp = vbase + (size_t)sg * CC;
#pragma unroll
      for (int db = 0; db < 8; ++db) {
        v8h t = *(const v8h*)(vp + db * 8);
#pragma unroll
        for (int j = 0; j < 8; ++j) vt[(db * 8 + j) * 32 + lane] = t[j];
      }
    } else {
#pragma unroll
      for (int d = 0; d < 64; ++d) vt[d * 32 + lane] = (half_t)0.f;
    }
    __syncthreads();
    const v16h pf = load_frag(pA, 32);
#pragma unroll
    for (int dt = 0; dt < 4; ++dt) {
      v16h vf = load_frag(&vt[dt * 16 * 32], 32);
      yacc[dt] = wmma32f16(pf, vf, yacc[dt]);
    }
    __syncthreads();
  }

  // Write Y (f16, (B,T,C) with head offset)
#pragma unroll
  for (int dt = 0; dt < 4; ++dt) {
#pragma unroll
    for (int a = 0; a < 8; ++a) {
      const int row = t0 + a + hi8;
      const int col = h * DD + dt * 16 + lc;
      y[((size_t)(b * TT + row)) * CC + col] = (half_t)yacc[dt][a];
    }
  }
}

// ---------------------------------------------------------------------------
// Host-side orchestration
// ---------------------------------------------------------------------------
extern "C" void kernel_launch(void* const* d_in, const int* in_sizes, int n_in,
                              void* d_out, int out_size, void* d_ws,
                              size_t ws_size, hipStream_t stream) {
  (void)in_sizes; (void)n_in; (void)out_size; (void)ws_size;
  // Inputs (setup_inputs order)
  const float* x     = (const float*)d_in[0];
  const float* ln1_g = (const float*)d_in[1];
  const float* ln1_b = (const float*)d_in[2];
  const float* ln2_g = (const float*)d_in[3];
  const float* ln2_b = (const float*)d_in[4];
  const float* Wq = (const float*)d_in[5];  const float* bq = (const float*)d_in[6];
  const float* Wk = (const float*)d_in[7];  const float* bk = (const float*)d_in[8];
  const float* Wv = (const float*)d_in[9];  const float* bv = (const float*)d_in[10];
  const float* Wp = (const float*)d_in[11]; const float* bp = (const float*)d_in[12];
  const float* W1 = (const float*)d_in[13]; const float* b1 = (const float*)d_in[14];
  const float* W2 = (const float*)d_in[15]; const float* b2 = (const float*)d_in[16];

  float* xout   = (float*)d_out;                       // B*T*C
  float* attout = (float*)d_out + (size_t)MM * CC;     // B*T*T

  // Workspace carve-out (256B aligned)
  char* w = (char*)d_ws;
  size_t off = 0;
  auto carve = [&](size_t bytes) -> char* {
    char* p = w + off;
    off += (bytes + 255) & ~(size_t)255;
    return p;
  };
  half_t* h16  = (half_t*)carve((size_t)MM * CC * 2);
  half_t* WqT  = (half_t*)carve((size_t)CC * CC * 2);   // C x C, N-major
  half_t* WkT  = (half_t*)carve((size_t)CC * CC * 2);
  half_t* WvT  = (half_t*)carve((size_t)CC * CC * 2);
  half_t* WpT  = (half_t*)carve((size_t)CC * CC * 2);
  half_t* W1T  = (half_t*)carve((size_t)CC * FFF * 2);  // FF x C
  half_t* W2T  = (half_t*)carve((size_t)FFF * CC * 2);  // C x FF
  half_t* q16  = (half_t*)carve((size_t)MM * CC * 2);
  half_t* k16  = (half_t*)carve((size_t)MM * CC * 2);
  half_t* v16  = (half_t*)carve((size_t)MM * CC * 2);
  half_t* y16  = (half_t*)carve((size_t)MM * CC * 2);
  float*  x1   = (float*) carve((size_t)MM * CC * 4);
  half_t* h2_16 = (half_t*)carve((size_t)MM * CC * 2);
  half_t* m16  = (half_t*)carve((size_t)MM * FFF * 2);

  // 0) zero att_mean accumulator in d_out
  k_zero_f32<<<8192, 256, 0, stream>>>(attout, (long long)MM * TT);

  // 1) weights fp32 -> f16, transposed to N-major (K-contiguous) for WMMA-B
  dim3 tb(32, 8);
  k_transpose_f32_f16<<<dim3(CC / 32, CC / 32), tb, 0, stream>>>(Wq, WqT, CC, CC);
  k_transpose_f32_f16<<<dim3(CC / 32, CC / 32), tb, 0, stream>>>(Wk, WkT, CC, CC);
  k_transpose_f32_f16<<<dim3(CC / 32, CC / 32), tb, 0, stream>>>(Wv, WvT, CC, CC);
  k_transpose_f32_f16<<<dim3(CC / 32, CC / 32), tb, 0, stream>>>(Wp, WpT, CC, CC);
  k_transpose_f32_f16<<<dim3(FFF / 32, CC / 32), tb, 0, stream>>>(W1, W1T, CC, FFF);
  k_transpose_f32_f16<<<dim3(CC / 32, FFF / 32), tb, 0, stream>>>(W2, W2T, FFF, CC);

  // 2) h = LN1(x)
  k_layernorm_f16<<<MM, 128, 0, stream>>>(x, ln1_g, ln1_b, h16);

  // 3) q,k,v = h @ W{q,k,v} + b
  dim3 gC(CC / 128, MM / 128);
  k_gemm_f16<<<gC, 256, 0, stream>>>(h16, WqT, bq, nullptr, nullptr, q16,
                                     MM, CC, CC, 0);
  k_gemm_f16<<<gC, 256, 0, stream>>>(h16, WkT, bk, nullptr, nullptr, k16,
                                     MM, CC, CC, 0);
  k_gemm_f16<<<gC, 256, 0, stream>>>(h16, WvT, bv, nullptr, nullptr, v16,
                                     MM, CC, CC, 0);

  // 4) attention (y + att_mean)
  k_attention<<<BB * HH * (TT / 16), 32, 0, stream>>>(q16, k16, v16, y16,
                                                      attout);

  // 5) x1 = x + (y @ Wp + bp)
  k_gemm_f16<<<gC, 256, 0, stream>>>(y16, WpT, bp, x, x1, nullptr,
                                     MM, CC, CC, 0);

  // 6) h2 = LN2(x1)
  k_layernorm_f16<<<MM, 128, 0, stream>>>(x1, ln2_g, ln2_b, h2_16);

  // 7) m = gelu(h2 @ W1 + b1)
  dim3 gF(FFF / 128, MM / 128);
  k_gemm_f16<<<gF, 256, 0, stream>>>(h2_16, W1T, b1, nullptr, nullptr, m16,
                                     MM, FFF, CC, 1);

  // 8) out = x1 + (m @ W2 + b2)
  k_gemm_f16<<<gC, 256, 0, stream>>>(m16, W2T, b2, x1, xout, nullptr,
                                     MM, CC, FFF, 0);
}
